// TransformerLayer_3539053052713
// MI455X (gfx1250) — compile-verified, matrix operand order checked by reading, not run on hardware
//
#include <hip/hip_runtime.h>
#include <hip/hip_bf16.h>
#include <math.h>

// ---------------- problem constants ----------------
constexpr int T_ = 2048, B_ = 2, D_ = 1024, H_ = 16, FF_ = 4096;
constexpr int HD_ = D_ / H_;          // 64
constexpr int N_  = T_ * B_;          // 4096 rows
constexpr int BH_ = B_ * H_;          // 32 heads
constexpr int TBP = 40;               // padded LDS row stride (halves): 32 data + 8 pad

typedef _Float16 v8h  __attribute__((ext_vector_type(8)));
typedef _Float16 v16h __attribute__((ext_vector_type(16)));
typedef float    v8f  __attribute__((ext_vector_type(8)));
typedef unsigned int u32;
typedef u32 u32x4 __attribute__((ext_vector_type(4)));
typedef int i32x4 __attribute__((ext_vector_type(4)));
typedef int i32x8 __attribute__((ext_vector_type(8)));

#if defined(__has_builtin)
#if __has_builtin(__builtin_amdgcn_tensor_load_to_lds) && __has_builtin(__builtin_amdgcn_s_wait_tensorcnt)
#define HAVE_TDM 1
#endif
#endif
#ifndef HAVE_TDM
#define HAVE_TDM 0
#endif

__device__ __forceinline__ v16h mk16(v8h lo, v8h hi) {
    v16h r;
#pragma unroll
    for (int i = 0; i < 8; ++i) { r[i] = lo[i]; r[i + 8] = hi[i]; }
    return r;
}

__device__ __forceinline__ v8f wmma_f16(v16h a, v16h b, v8f c) {
    // D = A(16x32 f16) * B(32x16 f16) + C(16x16 f32)
    return __builtin_amdgcn_wmma_f32_16x16x32_f16(false, a, false, b,
                                                  (short)0, c, false, false);
}

__device__ __forceinline__ float gelu_tanh(float x) {
    float x3 = x * x * x;
    return 0.5f * x * (1.0f + tanhf(0.7978845608028654f * (x + 0.044715f * x3)));
}

#if HAVE_TDM
// TDM: async-load a 128-row x 32-half tile (row stride K halves in memory) into LDS,
// padding each 64B row with 16B so the LDS row stride is TBP=40 halves.
// D# layout per CDNA5 ISA ch.8 (group0 128b, group1 256b; groups 2/3 zero => 2D).
__device__ __forceinline__ void tdm_tile_load(u32 lds_byte, const _Float16* gsrc, u32 k_halves) {
    unsigned long long ga = (unsigned long long)(uintptr_t)gsrc;
    u32x4 g0;
    g0[0] = 1u;                                   // count=1 (valid), no gather
    g0[1] = lds_byte;                             // lds_addr
    g0[2] = (u32)ga;                              // global_addr[31:0]
    g0[3] = (u32)((ga >> 32) & 0x01ffffffu) | (2u << 30);  // global_addr[56:32] | type=2
    i32x8 g1;
    // dw0: workgroup_mask=0 | data_size=1(2B)<<16 | pad_enable<<20 | pad_interval(16 DW)=3<<22 | pad_amount(4 DW)=3<<25
    g1[0] = (int)((1u << 16) | (1u << 20) | (3u << 22) | (3u << 25));
    g1[1] = (int)(32u << 16);                     // barrier_addr=0 | tensor_dim0 lo16 = 32
    g1[2] = (int)(128u << 16);                    // tensor_dim0 hi16 = 0 | tensor_dim1 lo16 = 128
    g1[3] = (int)(32u << 16);                     // tensor_dim1 hi16 = 0 | tile_dim0 = 32
    g1[4] = (int)(128u);                          // tile_dim1 = 128 | tile_dim2 = 0
    g1[5] = (int)k_halves;                        // tensor_dim0_stride[31:0] = K
    g1[6] = 0;                                    // stride0 hi | stride1 lo
    g1[7] = 0;                                    // stride1 hi
    i32x4 z4;
#pragma unroll
    for (int i = 0; i < 4; ++i) z4[i] = 0;
#if __clang_major__ >= 23
    i32x8 z8;
#pragma unroll
    for (int i = 0; i < 8; ++i) z8[i] = 0;
    __builtin_amdgcn_tensor_load_to_lds(g0, g1, z4, z4, z8, 0);
#else
    __builtin_amdgcn_tensor_load_to_lds(g0, g1, z4, z4, 0);
#endif
}
#endif

// ---------------- f32 -> f16 convert ----------------
__global__ void cvt_f32_f16(const float* __restrict__ in, _Float16* __restrict__ out, int n) {
    int i = blockIdx.x * 256 + threadIdx.x;
    if (i < n) out[i] = (_Float16)in[i];
}

// ---------------- tiled WMMA GEMM: out[M,O] = A[M,K] * W[O,K]^T + bias ----------------
// 128x128 block tile, double-buffered LDS, TDM async tile loads, 8 waves.
// EPI 0: f32 store.  EPI 1: tanh-GELU then f16 store.
template <int EPI>
__global__ __launch_bounds__(256) void gemm_kernel(
    const _Float16* __restrict__ A, const _Float16* __restrict__ W,
    const float* __restrict__ bias, void* __restrict__ outv,
    int K, int Ocols) {
    __shared__ _Float16 As[2][128 * TBP];
    __shared__ _Float16 Bs[2][128 * TBP];

    const int tid  = threadIdx.x;
    const int m0   = blockIdx.x * 128;
    const int n0   = blockIdx.y * 128;
    const int lane = tid & 31, wv = tid >> 5;
    const int lr   = lane & 15, lg = lane >> 4;
    const int koff = lg * 8;

    v8f acc[8];
#pragma unroll
    for (int j = 0; j < 8; ++j)
#pragma unroll
        for (int e = 0; e < 8; ++e) acc[j][e] = 0.0f;

    const int nsteps = K >> 5;   // K / 32

#if HAVE_TDM
    if (wv == 0) {
        tdm_tile_load((u32)(uintptr_t)&As[0][0], A + (size_t)m0 * K, (u32)K);
        tdm_tile_load((u32)(uintptr_t)&Bs[0][0], W + (size_t)n0 * K, (u32)K);
    }
#endif

    for (int ks = 0; ks < nsteps; ++ks) {
        const int kb = ks & 1;
#if HAVE_TDM
        if (wv == 0) {
            if (ks + 1 < nsteps) {
                // prefetch next stage into the other buffer (its last readers
                // finished at the end-of-iteration barrier of step ks-1)
                int kn = (ks + 1) << 5;
                tdm_tile_load((u32)(uintptr_t)&As[kb ^ 1][0], A + (size_t)m0 * K + kn, (u32)K);
                tdm_tile_load((u32)(uintptr_t)&Bs[kb ^ 1][0], W + (size_t)n0 * K + kn, (u32)K);
                __builtin_amdgcn_s_wait_tensorcnt(2);   // current stage's 2 TDM ops done
            } else {
                __builtin_amdgcn_s_wait_tensorcnt(0);
            }
        }
        __syncthreads();
#else
        const int k0 = ks << 5;
        __syncthreads();
#pragma unroll
        for (int c = tid; c < 512; c += 256) {          // 512 chunks of 8 halves
            int row = c >> 2, cc = c & 3;
            *(v8h*)&As[kb][row * TBP + cc * 8] =
                *(const v8h*)&A[(size_t)(m0 + row) * K + k0 + cc * 8];
            *(v8h*)&Bs[kb][row * TBP + cc * 8] =
                *(const v8h*)&W[(size_t)(n0 + row) * K + k0 + cc * 8];
        }
        __syncthreads();
#endif

        v16h af = mk16(*(v8h*)&As[kb][(wv * 16 + lr) * TBP + koff],
                       *(v8h*)&As[kb][(wv * 16 + lr) * TBP + koff + 16]);
#pragma unroll
        for (int j = 0; j < 8; ++j) {
            v16h bf = mk16(*(v8h*)&Bs[kb][(j * 16 + lr) * TBP + koff],
                           *(v8h*)&Bs[kb][(j * 16 + lr) * TBP + koff + 16]);
            acc[j] = wmma_f16(af, bf, acc[j]);
        }
        __syncthreads();   // all waves done reading buffer kb before it is refilled
    }

#pragma unroll
    for (int j = 0; j < 8; ++j) {
        int col = n0 + j * 16 + lr;
        float bv = bias[col];
#pragma unroll
        for (int r = 0; r < 8; ++r) {
            int rowg = m0 + wv * 16 + r + 8 * lg;
            float v = acc[j][r] + bv;
            if (EPI == 1) {
                v = gelu_tanh(v);
                ((_Float16*)outv)[(size_t)rowg * Ocols + col] = (_Float16)v;
            } else {
                ((float*)outv)[(size_t)rowg * Ocols + col] = v;
            }
        }
    }
}

// ---------------- repack qkv f32 -> per-head f16 (q scaled, v transposed) ----------------
__global__ void repack_qkv(const float* __restrict__ qkv,
                           _Float16* __restrict__ qh, _Float16* __restrict__ kh,
                           _Float16* __restrict__ vT) {
    int idx = blockIdx.x * 256 + threadIdx.x;
    if (idx >= BH_ * T_ * HD_) return;
    int head = idx / (T_ * HD_);
    int rem  = idx % (T_ * HD_);
    int t = rem / HD_, hd = rem % HD_;
    int b = head / H_, h = head % H_;
    size_t base = ((size_t)t * B_ + b) * (3 * D_) + h * HD_ + hd;
    qh[idx] = (_Float16)(qkv[base] * 0.125f);            // HD^-0.5 = 1/8
    kh[idx] = (_Float16)qkv[base + D_];
    vT[(size_t)head * HD_ * T_ + (size_t)hd * T_ + t] = (_Float16)qkv[base + 2 * D_];
}

// ---------------- flash attention (causal), one wave per 16 query rows ----------------
__global__ __launch_bounds__(256) void flash_kernel(
    const _Float16* __restrict__ qh, const _Float16* __restrict__ kh,
    const _Float16* __restrict__ vT, _Float16* __restrict__ attnh) {
    __shared__ _Float16 Pb[8][16 * TBP];

    const int tid  = threadIdx.x;
    const int lane = tid & 31, wv = tid >> 5;
    const int lr   = lane & 15, lg = lane >> 4;
    const int koff = lg * 8;
    const int head = blockIdx.y;
    const int qt   = blockIdx.x * 8 + wv;
    const int tq0  = qt * 16;

    const _Float16* qb = qh + (size_t)head * T_ * HD_;
    const _Float16* kb = kh + (size_t)head * T_ * HD_;
    const _Float16* vb = vT + (size_t)head * HD_ * T_;

    const _Float16* qrow = qb + (size_t)(tq0 + lr) * HD_;
    v16h aq0 = mk16(*(const v8h*)(qrow + koff),      *(const v8h*)(qrow + koff + 16));
    v16h aq1 = mk16(*(const v8h*)(qrow + 32 + koff), *(const v8h*)(qrow + 32 + koff + 16));

    v8f o[4];
    v8f zf;
#pragma unroll
    for (int e = 0; e < 8; ++e) zf[e] = 0.0f;
#pragma unroll
    for (int nt = 0; nt < 4; ++nt) o[nt] = zf;
    float mrow[8], lrow[8];
#pragma unroll
    for (int r = 0; r < 8; ++r) { mrow[r] = -1e30f; lrow[r] = 0.0f; }

    _Float16* pb = &Pb[wv][0];

    for (int kt = 0; kt <= tq0 + 15; kt += 32) {
        v8f s[2];
#pragma unroll
        for (int jt = 0; jt < 2; ++jt) {
            const _Float16* krow = kb + (size_t)(kt + jt * 16 + lr) * HD_;
            v16h bk0 = mk16(*(const v8h*)(krow + koff),      *(const v8h*)(krow + koff + 16));
            v16h bk1 = mk16(*(const v8h*)(krow + 32 + koff), *(const v8h*)(krow + 32 + koff + 16));
            v8f t = wmma_f16(aq0, bk0, zf);
            s[jt]  = wmma_f16(aq1, bk1, t);
        }
        // online softmax per row (C layout: row = r + 8*lg, col = lr)
#pragma unroll
        for (int r = 0; r < 8; ++r) {
            int tq = tq0 + r + 8 * lg;
            float x0 = (kt + lr      > tq) ? -1e30f : s[0][r];
            float x1 = (kt + 16 + lr > tq) ? -1e30f : s[1][r];
            float cmax = fmaxf(x0, x1);
#pragma unroll
            for (int msk = 1; msk < 16; msk <<= 1)
                cmax = fmaxf(cmax, __shfl_xor(cmax, msk, 32));
            float mnew = fmaxf(mrow[r], cmax);
            float sc   = __expf(mrow[r] - mnew);
            float p0   = __expf(x0 - mnew);
            float p1   = __expf(x1 - mnew);
            float ps   = p0 + p1;
#pragma unroll
            for (int msk = 1; msk < 16; msk <<= 1)
                ps += __shfl_xor(ps, msk, 32);
            lrow[r] = lrow[r] * sc + ps;
            mrow[r] = mnew;
#pragma unroll
            for (int nt = 0; nt < 4; ++nt) o[nt][r] *= sc;
            pb[(r + 8 * lg) * TBP + lr]      = (_Float16)p0;
            pb[(r + 8 * lg) * TBP + 16 + lr] = (_Float16)p1;
        }
        asm volatile("" ::: "memory");  // keep P stores before fragment reload
        v16h ap = mk16(*(v8h*)&pb[lr * TBP + koff], *(v8h*)&pb[lr * TBP + koff + 16]);
#pragma unroll
        for (int nt = 0; nt < 4; ++nt) {
            const _Float16* vrow = vb + (size_t)(nt * 16 + lr) * T_ + kt;
            v16h bv = mk16(*(const v8h*)(vrow + koff), *(const v8h*)(vrow + koff + 16));
            o[nt] = wmma_f16(ap, bv, o[nt]);
        }
    }

    const int b = head / H_, h = head % H_;
#pragma unroll
    for (int r = 0; r < 8; ++r) {
        float inv = 1.0f / lrow[r];
        int tq = tq0 + r + 8 * lg;
        size_t rowoff = ((size_t)tq * B_ + b) * D_ + h * HD_;
#pragma unroll
        for (int nt = 0; nt < 4; ++nt)
            attnh[rowoff + nt * 16 + lr] = (_Float16)(o[nt][r] * inv);
    }
}

// ---------------- layernorm of (a + res): f32 out + optional f16 out ----------------
__global__ __launch_bounds__(256) void ln_kernel(
    const float* __restrict__ a, const float* __restrict__ res,
    const float* __restrict__ g, const float* __restrict__ bta,
    float* __restrict__ out32, _Float16* __restrict__ out16) {
    __shared__ float red[256];
    const int row = blockIdx.x, tid = threadIdx.x;
    const float* pa = a   + (size_t)row * D_;
    const float* pr = res + (size_t)row * D_;

    float v[4], s = 0.0f;
#pragma unroll
    for (int i = 0; i < 4; ++i) { v[i] = pa[tid + i * 256] + pr[tid + i * 256]; s += v[i]; }
    red[tid] = s; __syncthreads();
    for (int st = 128; st > 0; st >>= 1) { if (tid < st) red[tid] += red[tid + st]; __syncthreads(); }
    float mu = red[0] * (1.0f / D_);
    __syncthreads();

    float q = 0.0f;
#pragma unroll
    for (int i = 0; i < 4; ++i) { float d = v[i] - mu; q += d * d; }
    red[tid] = q; __syncthreads();
    for (int st = 128; st > 0; st >>= 1) { if (tid < st) red[tid] += red[tid + st]; __syncthreads(); }
    float inv = rsqrtf(red[0] * (1.0f / D_) + 1e-5f);

#pragma unroll
    for (int i = 0; i < 4; ++i) {
        int c = tid + i * 256;
        float y = (v[i] - mu) * inv * g[c] + bta[c];
        out32[(size_t)row * D_ + c] = y;
        if (out16) out16[(size_t)row * D_ + c] = (_Float16)y;
    }
}

// ---------------- host orchestration ----------------
extern "C" void kernel_launch(void* const* d_in, const int* in_sizes, int n_in,
                              void* d_out, int out_size, void* d_ws, size_t ws_size,
                              hipStream_t stream) {
    const float* x        = (const float*)d_in[0];
    // d_in[1] = attn_mask (bool) — causal mask is computed analytically
    const float* in_w     = (const float*)d_in[2];
    const float* in_b     = (const float*)d_in[3];
    const float* out_w    = (const float*)d_in[4];
    const float* out_b    = (const float*)d_in[5];
    const float* fc1_w    = (const float*)d_in[6];
    const float* fc1_b    = (const float*)d_in[7];
    const float* fc2_w    = (const float*)d_in[8];
    const float* fc2_b    = (const float*)d_in[9];
    const float* ln1_g    = (const float*)d_in[10];
    const float* ln1_b    = (const float*)d_in[11];
    const float* ln2_g    = (const float*)d_in[12];
    const float* ln2_b    = (const float*)d_in[13];
    float* out            = (float*)d_out;

    char* ws = (char*)d_ws;
    size_t off = 0;
    auto alloc = [&](size_t bytes) -> void* {
        void* p = ws + off;
        off = (off + bytes + 255) & ~(size_t)255;
        return p;
    };

    _Float16* xh     = (_Float16*)alloc((size_t)N_ * D_ * 2);
    _Float16* wqkvh  = (_Float16*)alloc((size_t)3 * D_ * D_ * 2);
    _Float16* wouth  = (_Float16*)alloc((size_t)D_ * D_ * 2);
    _Float16* wfc1h  = (_Float16*)alloc((size_t)FF_ * D_ * 2);
    _Float16* wfc2h  = (_Float16*)alloc((size_t)D_ * FF_ * 2);
    float*    qkv32  = (float*)   alloc((size_t)N_ * 3 * D_ * 4);
    _Float16* qhp    = (_Float16*)alloc((size_t)BH_ * T_ * HD_ * 2);
    _Float16* khp    = (_Float16*)alloc((size_t)BH_ * T_ * HD_ * 2);
    _Float16* vTp    = (_Float16*)alloc((size_t)BH_ * HD_ * T_ * 2);
    _Float16* attnh  = (_Float16*)alloc((size_t)N_ * D_ * 2);
    float*    aproj  = (float*)   alloc((size_t)N_ * D_ * 4);
    float*    hres   = (float*)   alloc((size_t)N_ * D_ * 4);
    _Float16* hh     = (_Float16*)alloc((size_t)N_ * D_ * 2);
    _Float16* ff1h   = (_Float16*)alloc((size_t)N_ * FF_ * 2);
    float*    ff2    = (float*)   alloc((size_t)N_ * D_ * 4);
    (void)ws_size; (void)in_sizes; (void)n_in; (void)out_size;

    auto cvt = [&](const float* src, _Float16* dst, int n) {
        cvt_f32_f16<<<(n + 255) / 256, 256, 0, stream>>>(src, dst, n);
    };
    cvt(x,     xh,    N_ * D_);
    cvt(in_w,  wqkvh, 3 * D_ * D_);
    cvt(out_w, wouth, D_ * D_);
    cvt(fc1_w, wfc1h, FF_ * D_);
    cvt(fc2_w, wfc2h, D_ * FF_);

    // 1) QKV projection: [N,3D] f32
    gemm_kernel<0><<<dim3(N_ / 128, (3 * D_) / 128), 256, 0, stream>>>(
        xh, wqkvh, in_b, qkv32, D_, 3 * D_);

    // 2) repack to per-head q/k/vT (q pre-scaled)
    {
        int total = BH_ * T_ * HD_;
        repack_qkv<<<(total + 255) / 256, 256, 0, stream>>>(qkv32, qhp, khp, vTp);
    }

    // 3) causal flash attention -> attnh f16 [N, D]
    flash_kernel<<<dim3(T_ / 128, BH_), 256, 0, stream>>>(qhp, khp, vTp, attnh);

    // 4) output projection
    gemm_kernel<0><<<dim3(N_ / 128, D_ / 128), 256, 0, stream>>>(
        attnh, wouth, out_b, aproj, D_, D_);

    // 5) h = LN1(x + attn) -> f32 residual + f16 operand
    ln_kernel<<<N_, 256, 0, stream>>>(x, aproj, ln1_g, ln1_b, hres, hh);

    // 6) FC1 + GELU -> f16 [N, FF]
    gemm_kernel<1><<<dim3(N_ / 128, FF_ / 128), 256, 0, stream>>>(
        hh, wfc1h, fc1_b, ff1h, D_, FF_);

    // 7) FC2 -> f32 [N, D]
    gemm_kernel<0><<<dim3(N_ / 128, D_ / 128), 256, 0, stream>>>(
        ff1h, wfc2h, fc2_b, ff2, FF_, D_);

    // 8) out = LN2(h + ff) -> d_out
    ln_kernel<<<N_, 256, 0, stream>>>(hres, ff2, ln2_g, ln2_b, out, nullptr);
}
